// MultiHeadAttention_52664888984171
// MI455X (gfx1250) — matrix operand order repaired and checked by reading, not compiled, not run on hardware
//
#include <hip/hip_runtime.h>
#include <math.h>

// MHA forward for B=4, S=2048, D=1024, H=16, HD=64 on gfx1250 (wave32, WMMA).
// Pipeline: f32->f16 convert, Wt transpose, LDS-tiled WMMA projection GEMMs
// (async global->LDS staging), flash-attention (online softmax, WMMA QK^T/PV),
// LDS-tiled WMMA output GEMM.

typedef _Float16 h16;
typedef __attribute__((ext_vector_type(16))) _Float16 v16h;
typedef __attribute__((ext_vector_type(8)))  _Float16 v8h;
typedef __attribute__((ext_vector_type(8)))  float    v8f;
typedef __attribute__((ext_vector_type(4)))  int      v4i;

#define B_  4
#define S_  2048
#define D_  1024
#define H_  16
#define HD_ 64

#define BM 128
#define BN 128
#define BK 32

#if __has_builtin(__builtin_amdgcn_global_load_async_to_lds_b128)
#define HAVE_ASYNC_LDS 1
#else
#define HAVE_ASYNC_LDS 0
#endif

static __device__ __forceinline__ void async_wait0() {
#if __has_builtin(__builtin_amdgcn_s_wait_asynccnt)
    __builtin_amdgcn_s_wait_asynccnt(0);
#else
    asm volatile("s_wait_asynccnt 0" ::: "memory");
#endif
}

static __device__ __forceinline__ void copy16B_to_lds(const h16* g, h16* l) {
#if HAVE_ASYNC_LDS
    __builtin_amdgcn_global_load_async_to_lds_b128(
        (__attribute__((address_space(1))) v4i*)(g),
        (__attribute__((address_space(3))) v4i*)(l), 0, 0);
#else
    *(v8h*)l = *(const v8h*)g;
#endif
}

// ---------------------------------------------------------------- converts
__global__ __launch_bounds__(256) void f32_to_f16_kernel(const float* __restrict__ in,
                                                         h16* __restrict__ out, int n) {
    int i = blockIdx.x * blockDim.x + threadIdx.x;
    if (i < n) out[i] = (h16)in[i];
}

// W is (K,N) row-major f32; produce Wt (N,K) row-major f16.
__global__ __launch_bounds__(256) void w_transpose_f16(const float* __restrict__ W,
                                                       h16* __restrict__ Wt, int K, int N) {
    int i = blockIdx.x * blockDim.x + threadIdx.x;
    if (i >= K * N) return;
    int k = i / N, n = i % N;
    Wt[(size_t)n * K + k] = (h16)W[(size_t)k * N + n];
}

// ---------------------------------------------------------------- tiled GEMM
// C[M,N] = A[M,K] @ Bt[N,K]^T + bias.  8-wave block computes a 128x128 tile;
// per K-step, A/B 32-wide slabs are staged to LDS (async global->LDS when
// available), then each wave does 8 WMMAs (one A-frag reused over 8 B-frags).
// mode 0: f16 out, head-split  [b,h,s,hd]
// mode 1: f16 out, head-split transposed [b,h,hd,s]   (for V)
// mode 2: f32 out, row-major [M,N]                    (final projection)
__global__ __launch_bounds__(256) void gemm_wmma_tiled(const h16* __restrict__ A,
                                                       const h16* __restrict__ Bt,
                                                       const float* __restrict__ bias,
                                                       h16* __restrict__ outH,
                                                       float* __restrict__ outF,
                                                       int M, int N, int K, int mode) {
    __shared__ h16 As[BM * BK];   // [row][k]  8 KB
    __shared__ h16 Bs[BN * BK];   // [col][k]  8 KB
    int tid = threadIdx.x;
    int w = tid >> 5, lane = tid & 31;
    int mr = lane & 15, hf = lane >> 4;
    int nb = N / BN;
    int bm = blockIdx.x / nb, bn = blockIdx.x % nb;
    int gm0 = bm * BM, gn0 = bn * BN;

    v8f acc[8];
#pragma unroll
    for (int t = 0; t < 8; ++t) acc[t] = (v8f){};

    int srow = tid >> 2;          // 0..63
    int skq  = (tid & 3) * 8;     // k-offset within slab: 0,8,16,24

    const h16* ga = A  + (size_t)(gm0 + srow) * K + skq;
    const h16* gb = Bt + (size_t)(gn0 + srow) * K + skq;

    for (int k0 = 0; k0 < K; k0 += BK) {
        // ---- stage A (128x32) and B (128x32) slabs into LDS
        copy16B_to_lds(ga + k0,                     As + srow * BK + skq);
        copy16B_to_lds(ga + (size_t)64 * K + k0,    As + (srow + 64) * BK + skq);
        copy16B_to_lds(gb + k0,                     Bs + srow * BK + skq);
        copy16B_to_lds(gb + (size_t)64 * K + k0,    Bs + (srow + 64) * BK + skq);
#if HAVE_ASYNC_LDS
        async_wait0();
#endif
        __syncthreads();

        // ---- one A-frag per wave, 8 B-frags, 8 WMMAs
        v16h a;
        const h16* ap = As + (w * 16 + mr) * BK + hf * 16;
#pragma unroll
        for (int j = 0; j < 16; ++j) { a[j] = ap[j]; }
#pragma unroll
        for (int ct = 0; ct < 8; ++ct) {
            v16h b;
            const h16* bp = Bs + (ct * 16 + mr) * BK + hf * 16;
#pragma unroll
            for (int j = 0; j < 16; ++j) b[j] = bp[j];
            acc[ct] = __builtin_amdgcn_wmma_f32_16x16x32_f16(false, a, false, b,
                                                             (short)0, acc[ct], false, false);
        }
        __syncthreads();
    }

    // ---- epilogue
#pragma unroll
    for (int ct = 0; ct < 8; ++ct) {
        int col = gn0 + ct * 16 + mr;
        float bc = bias ? bias[col] : 0.0f;
#pragma unroll
        for (int r = 0; r < 8; ++r) {
            int row = gm0 + w * 16 + r + 8 * hf;
            float v = acc[ct][r] + bc;
            if (mode == 2) {
                outF[(size_t)row * N + col] = v;
            } else {
                int bb = row >> 11, ss = row & (S_ - 1);     // row = b*S + s
                int hh = col >> 6,  hd = col & (HD_ - 1);    // col = h*HD + hd
                if (mode == 0)
                    outH[(((size_t)bb * H_ + hh) * S_ + ss) * HD_ + hd] = (h16)v;
                else
                    outH[(((size_t)bb * H_ + hh) * HD_ + hd) * S_ + ss] = (h16)v;
            }
        }
    }
}

// ---------------------------------------------------------------- flash attention
// One wave per (b,h, 16-row query tile). Online softmax over causal keys in
// chunks of 32. Scores via 2x WMMA (K=HD=64) per 16-key sub-tile, P bounced
// through LDS to A-layout, PV via 4x WMMA (N tiles of 16 over HD=64).
__global__ __launch_bounds__(256) void attn_kernel(const h16* __restrict__ Qh,
                                                   const h16* __restrict__ Kh,
                                                   const h16* __restrict__ Vt,
                                                   h16* __restrict__ Ctx) {
    __shared__ h16 pbuf[8][16 * 32];   // per-wave 16x32 f16 P tile (8 KB)
    int w = threadIdx.x >> 5;
    int wave = blockIdx.x * 8 + w;
    int qt = wave & (S_ / 16 - 1);     // 0..127
    int bh = wave >> 7;                // b*H + h
    int b = bh >> 4, h = bh & (H_ - 1);
    int qbase = qt << 4;
    int lane = threadIdx.x & 31;
    int mr = lane & 15, hf = lane >> 4;
    const float scale = 0.125f;        // 1/sqrt(HD)

    // Q A-fragments for the two 32-wide K chunks of HD=64
    const h16* qp = Qh + ((size_t)bh * S_ + qbase + mr) * HD_ + hf * 16;
    v16h qa0, qa1;
#pragma unroll
    for (int j = 0; j < 16; ++j) {
        qa0[j] = qp[j];
        qa1[j] = qp[32 + j];
    }

    float mrow[8], lrow[8];
    v8f o0 = {}, o1 = {}, o2 = {}, o3 = {};
#pragma unroll
    for (int r = 0; r < 8; ++r) { mrow[r] = -1e30f; lrow[r] = 0.0f; }

    h16* pw = &pbuf[w][0];

    for (int kb = 0; kb <= qbase + 15; kb += 32) {
        // ---- scores for two 16-key sub-tiles
        v8f s0 = {}, s1 = {};
        {
            const h16* kp = Kh + ((size_t)bh * S_ + kb + mr) * HD_ + hf * 16;
            v16h b0, b1;
#pragma unroll
            for (int j = 0; j < 16; ++j) { b0[j] = kp[j]; b1[j] = kp[32 + j]; }
            s0 = __builtin_amdgcn_wmma_f32_16x16x32_f16(false, qa0, false, b0, (short)0, s0, false, false);
            s0 = __builtin_amdgcn_wmma_f32_16x16x32_f16(false, qa1, false, b1, (short)0, s0, false, false);
        }
        {
            const h16* kp = Kh + ((size_t)bh * S_ + kb + 16 + mr) * HD_ + hf * 16;
            v16h b0, b1;
#pragma unroll
            for (int j = 0; j < 16; ++j) { b0[j] = kp[j]; b1[j] = kp[32 + j]; }
            s1 = __builtin_amdgcn_wmma_f32_16x16x32_f16(false, qa0, false, b0, (short)0, s1, false, false);
            s1 = __builtin_amdgcn_wmma_f32_16x16x32_f16(false, qa1, false, b1, (short)0, s1, false, false);
        }

        // ---- causal mask + scale, online softmax update
        float mnew[8], corr[8];
#pragma unroll
        for (int r = 0; r < 8; ++r) {
            int row = qbase + r + 8 * hf;
            s0[r] = ((kb + mr)      <= row) ? s0[r] * scale : -1e30f;
            s1[r] = ((kb + 16 + mr) <= row) ? s1[r] * scale : -1e30f;
            float t = fmaxf(s0[r], s1[r]);
            t = fmaxf(t, __shfl_xor(t, 1, 32));
            t = fmaxf(t, __shfl_xor(t, 2, 32));
            t = fmaxf(t, __shfl_xor(t, 4, 32));
            t = fmaxf(t, __shfl_xor(t, 8, 32));
            mnew[r] = fmaxf(mrow[r], t);
            corr[r] = __expf(mrow[r] - mnew[r]);
        }
#pragma unroll
        for (int r = 0; r < 8; ++r) {
            float p0 = __expf(s0[r] - mnew[r]);
            float p1 = __expf(s1[r] - mnew[r]);
            float u = p0 + p1;
            u += __shfl_xor(u, 1, 32);
            u += __shfl_xor(u, 2, 32);
            u += __shfl_xor(u, 4, 32);
            u += __shfl_xor(u, 8, 32);
            lrow[r] = lrow[r] * corr[r] + u;
            mrow[r] = mnew[r];
            o0[r] *= corr[r]; o1[r] *= corr[r]; o2[r] *= corr[r]; o3[r] *= corr[r];
            int li = (r + 8 * hf) * 32;          // C-layout -> LDS row-major 16x32
            pw[li + mr]      = (h16)p0;
            pw[li + 16 + mr] = (h16)p1;
        }
        asm volatile("s_wait_dscnt 0" ::: "memory");   // per-wave DS ordering

        // ---- reload P in A-layout, accumulate PV (4 N-tiles of 16 over HD)
        v16h pa;
        const h16* pp = pw + mr * 32 + hf * 16;
#pragma unroll
        for (int j = 0; j < 16; ++j) { pa[j] = pp[j]; }

        const h16* vp = Vt + ((size_t)bh * HD_ + mr) * S_ + kb + hf * 16;
        {
            v16h vb;
#pragma unroll
            for (int j = 0; j < 16; ++j) vb[j] = vp[j];
            o0 = __builtin_amdgcn_wmma_f32_16x16x32_f16(false, pa, false, vb, (short)0, o0, false, false);
        }
        {
            v16h vb;
#pragma unroll
            for (int j = 0; j < 16; ++j) vb[j] = vp[(size_t)16 * S_ + j];
            o1 = __builtin_amdgcn_wmma_f32_16x16x32_f16(false, pa, false, vb, (short)0, o1, false, false);
        }
        {
            v16h vb;
#pragma unroll
            for (int j = 0; j < 16; ++j) vb[j] = vp[(size_t)32 * S_ + j];
            o2 = __builtin_amdgcn_wmma_f32_16x16x32_f16(false, pa, false, vb, (short)0, o2, false, false);
        }
        {
            v16h vb;
#pragma unroll
            for (int j = 0; j < 16; ++j) vb[j] = vp[(size_t)48 * S_ + j];
            o3 = __builtin_amdgcn_wmma_f32_16x16x32_f16(false, pa, false, vb, (short)0, o3, false, false);
        }
    }

    // ---- normalize and write merged-head context [b, s, h*HD + hd] (f16)
#pragma unroll
    for (int r = 0; r < 8; ++r) {
        float inv = 1.0f / lrow[r];
        int row = qbase + r + 8 * hf;
        size_t base = ((size_t)b * S_ + row) * D_ + h * HD_;
        Ctx[base +  0 + mr] = (h16)(o0[r] * inv);
        Ctx[base + 16 + mr] = (h16)(o1[r] * inv);
        Ctx[base + 32 + mr] = (h16)(o2[r] * inv);
        Ctx[base + 48 + mr] = (h16)(o3[r] * inv);
    }
}

// ---------------------------------------------------------------- launch
extern "C" void kernel_launch(void* const* d_in, const int* in_sizes, int n_in,
                              void* d_out, int out_size, void* d_ws, size_t ws_size,
                              hipStream_t stream) {
    const float* q  = (const float*)d_in[0];
    const float* k  = (const float*)d_in[1];
    const float* v  = (const float*)d_in[2];
    // d_in[3] = mask: causal tril, implemented analytically in attn_kernel
    const float* Wq = (const float*)d_in[4];  const float* bq = (const float*)d_in[5];
    const float* Wk = (const float*)d_in[6];  const float* bk = (const float*)d_in[7];
    const float* Wv = (const float*)d_in[8];  const float* bv = (const float*)d_in[9];
    const float* Wo = (const float*)d_in[10]; const float* bo = (const float*)d_in[11];
    float* out = (float*)d_out;

    const size_t NT = (size_t)B_ * S_;     // 8192 tokens
    const size_t XE = NT * D_;             // 8,388,608
    const size_t WE = (size_t)D_ * D_;     // 1,048,576

    h16* ws  = (h16*)d_ws;
    h16* Xq  = ws;        h16* Xk  = Xq  + XE;  h16* Xv  = Xk  + XE;
    h16* WqT = Xv + XE;   h16* WkT = WqT + WE;  h16* WvT = WkT + WE;  h16* WoT = WvT + WE;
    h16* Qh  = WoT + WE;  h16* Kh  = Qh  + XE;  h16* Vt  = Kh  + XE;  h16* Ctx = Vt + XE;

    dim3 blk(256);
    int nx = (int)XE;
    f32_to_f16_kernel<<<(nx + 255) / 256, blk, 0, stream>>>(q, Xq, nx);
    f32_to_f16_kernel<<<(nx + 255) / 256, blk, 0, stream>>>(k, Xk, nx);
    f32_to_f16_kernel<<<(nx + 255) / 256, blk, 0, stream>>>(v, Xv, nx);

    int nw = (int)WE;
    w_transpose_f16<<<(nw + 255) / 256, blk, 0, stream>>>(Wq, WqT, D_, D_);
    w_transpose_f16<<<(nw + 255) / 256, blk, 0, stream>>>(Wk, WkT, D_, D_);
    w_transpose_f16<<<(nw + 255) / 256, blk, 0, stream>>>(Wv, WvT, D_, D_);
    w_transpose_f16<<<(nw + 255) / 256, blk, 0, stream>>>(Wo, WoT, D_, D_);

    int gblocks = (int)((NT / BM) * (D_ / BN));   // 64 * 8 = 512 blocks
    gemm_wmma_tiled<<<gblocks, 256, 0, stream>>>(Xq, WqT, bq, Qh, nullptr, (int)NT, D_, D_, 0);
    gemm_wmma_tiled<<<gblocks, 256, 0, stream>>>(Xk, WkT, bk, Kh, nullptr, (int)NT, D_, D_, 0);
    gemm_wmma_tiled<<<gblocks, 256, 0, stream>>>(Xv, WvT, bv, Vt, nullptr, (int)NT, D_, D_, 1);

    attn_kernel<<<(B_ * H_ * (S_ / 16)) / 8, 256, 0, stream>>>(Qh, Kh, Vt, Ctx);

    gemm_wmma_tiled<<<gblocks, 256, 0, stream>>>(Ctx, WoT, bo, nullptr, out, (int)NT, D_, D_, 2);
}
